// CrossAttentionBlock_1683627180795
// MI455X (gfx1250) — matrix operand order repaired and checked by reading, not compiled, not run on hardware
//
#include <hip/hip_runtime.h>

// ---- problem constants ----
#define DMODEL 1024
#define NHEAD  16
#define HD     64
#define LQ     2048
#define LKV    3072
#define NB     2
#define MQ     (NB*LQ)    // 4096 query rows
#define MKV    (NB*LKV)   // 6144 kv rows

typedef __bf16 bf16;
typedef __attribute__((ext_vector_type(16))) __bf16 v16bf;
typedef __attribute__((ext_vector_type(8)))  __bf16 v8bf;
typedef __attribute__((ext_vector_type(8)))  float  v8f;

union A16 { v16bf v; v8bf h[2]; };

static __device__ __forceinline__ v8f wmma_bf16(v16bf a, v16bf b, v8f c) {
  return __builtin_amdgcn_wmma_f32_16x16x32_bf16(false, a, false, b, (short)0, c, false, false);
}

// ---------------- LayerNorm: one wave per 1024-elem row, bf16 out ----------------
__global__ void ln_kernel(const float* __restrict__ x, const float* __restrict__ g,
                          const float* __restrict__ bb, bf16* __restrict__ out, int rows) {
  int wave = (blockIdx.x * blockDim.x + threadIdx.x) >> 5;
  int lane = threadIdx.x & 31;
  if (wave >= rows) return;
  const float* xr = x + (size_t)wave * DMODEL;
  float vals[32];
  float s = 0.f, ss = 0.f;
  #pragma unroll
  for (int i = 0; i < 32; ++i) {
    float t = xr[lane + 32*i];
    vals[i] = t; s += t; ss += t*t;
  }
  #pragma unroll
  for (int m = 1; m <= 16; m <<= 1) { s += __shfl_xor(s, m, 32); ss += __shfl_xor(ss, m, 32); }
  float mean = s * (1.0f/DMODEL);
  float var  = ss * (1.0f/DMODEL) - mean*mean;
  float rstd = rsqrtf(var + 1e-5f);
  bf16* orow = out + (size_t)wave * DMODEL;
  #pragma unroll
  for (int i = 0; i < 32; ++i) {
    int c = lane + 32*i;
    orow[c] = (bf16)((vals[i] - mean) * rstd * g[c] + bb[c]);
  }
}

// ---------------- W[n][k] -> WT[k][n] (bf16), for WMMA B operand (K-major) ----------------
__global__ void wtrans_kernel(const float* __restrict__ W, bf16* __restrict__ WT) {
  int idx = blockIdx.x * blockDim.x + threadIdx.x;       // 2^18 threads
  #pragma unroll
  for (int r = 0; r < 4; ++r) {
    int e = idx + r * (1 << 18);                          // coalesced writes over WT
    int k = e >> 10, n = e & 1023;
    WT[e] = (bf16)W[n * DMODEL + k];
  }
}

// ---------------- GEMM: out[M x 1024] = A(bf16) @ WT(bf16) + bias ----------------
// mode 0: bf16 natural [M][D] ; mode 1: per-head kT layout [b][h][d][key] (bf16, packed stores)
// mode 2: f32 out = acc + bias + resid  (final projection + residual)
__global__ void gemm_kernel(const bf16* __restrict__ A, const bf16* __restrict__ BT,
                            const float* __restrict__ bias, int M, int mode,
                            bf16* __restrict__ outb, float* __restrict__ outf,
                            const float* __restrict__ resid) {
  int task = (int)((blockIdx.x * blockDim.x + threadIdx.x) >> 5);  // one wave = 16x64 tile
  int lane = threadIdx.x & 31;
  const int nstrips = DMODEL / 64;        // 16
  int mt = task / nstrips;
  int ns = task % nstrips;
  int m0 = mt * 16, n0 = ns * 64;
  if (m0 >= M) return;                    // uniform per wave
  int mrow = lane & 15, half = lane >> 4, base = half * 8;

  v8f acc[4] = {};
  const bf16* arow = A + (size_t)(m0 + mrow) * DMODEL;
  for (int k0 = 0; k0 < DMODEL; k0 += 32) {
    A16 a;
    a.h[0] = *(const v8bf*)(arow + k0 + 8*half);         // K = 8h..8h+7
    a.h[1] = *(const v8bf*)(arow + k0 + 16 + 8*half);    // K = 16+8h..
    const bf16* brow = BT + (size_t)k0 * DMODEL + n0 + mrow;   // B: lane = N column, elems = K (table order)
    #pragma unroll
    for (int j = 0; j < 4; ++j) {
      v16bf b;
      #pragma unroll
      for (int s = 0; s < 16; ++s) b[s] = brow[(size_t)((s & 7) + 8*half + 16*(s >> 3)) * DMODEL + 16*j];
      acc[j] = wmma_bf16(a.v, b, acc[j]);
    }
  }

  #pragma unroll
  for (int j = 0; j < 4; ++j) {
    int n = n0 + 16*j + mrow;
    float bv = bias[n];
    if (mode == 0) {
      #pragma unroll
      for (int i = 0; i < 8; ++i)
        outb[(size_t)(m0 + base + i) * DMODEL + n] = (bf16)(acc[j][i] + bv);
    } else if (mode == 1) {
      int bidx = m0 / LKV;                       // tile never crosses batch (3072 % 16 == 0)
      int keyb = (m0 - bidx * LKV) + base;
      int hh = n / HD, dd = n % HD;
      bf16* dst = outb + ((size_t)(bidx * NHEAD + hh) * HD + dd) * LKV + keyb;
      union { v8bf v; bf16 e[8]; } pk;
      #pragma unroll
      for (int i = 0; i < 8; ++i) pk.e[i] = (bf16)(acc[j][i] + bv);
      *(v8bf*)dst = pk.v;                        // 8 consecutive keys, 16B store
    } else {
      #pragma unroll
      for (int i = 0; i < 8; ++i) {
        size_t o = (size_t)(m0 + base + i) * DMODEL + n;
        outf[o] = acc[j][i] + bv + resid[o];
      }
    }
  }
}

// ---------------- flash attention: one wave per (b, h, 16-query tile) ----------------
__global__ void attn_kernel(const bf16* __restrict__ qp, const bf16* __restrict__ kT,
                            const bf16* __restrict__ vp, bf16* __restrict__ outp) {
  __shared__ __align__(32) bf16 plds[8][16 * 32];    // per-wave 16x32 P tile
  int wid  = threadIdx.x >> 5;
  int lane = threadIdx.x & 31;
  int task = blockIdx.x * 8 + wid;                   // 0..4095
  int qt = task & 127;                               // LQ/16 = 128
  int bh = task >> 7;
  int h = bh & 15, b = bh >> 4;
  int mrow = lane & 15, half = lane >> 4, base = half * 8;

  // Q tile (16 x 64), A-operand layout, loaded once
  const bf16* qrow = qp + ((size_t)(b * LQ) + qt * 16 + mrow) * DMODEL + h * HD;
  A16 aql, aqh;
  aql.h[0] = *(const v8bf*)(qrow + 8*half);
  aql.h[1] = *(const v8bf*)(qrow + 16 + 8*half);
  aqh.h[0] = *(const v8bf*)(qrow + 32 + 8*half);
  aqh.h[1] = *(const v8bf*)(qrow + 48 + 8*half);

  const bf16* kThead = kT + (size_t)(b * NHEAD + h) * HD * LKV;
  const bf16* vhead  = vp + (size_t)b * LKV * DMODEL + h * HD;
  bf16* myp = plds[wid];

  float mrun[8], lrun[8];
  #pragma unroll
  for (int i = 0; i < 8; ++i) { mrun[i] = -1e30f; lrun[i] = 0.f; }
  v8f acc[4] = {};

  for (int key0 = 0; key0 < LKV; key0 += 32) {
    // scores: two 16x16 tiles (keys key0..+15, key0+16..+31), inner dim d=64
    const bf16* kb0 = kThead + key0 + mrow;                      // B column key = mrow, elems d (table order)
    const bf16* kb1 = kThead + (size_t)32 * LKV + key0 + mrow;   // same, d = 32+..
    v16bf b00, b10, b01, b11;
    #pragma unroll
    for (int s = 0; s < 16; ++s) {
      size_t kk = (size_t)((s & 7) + 8*half + 16*(s >> 3)) * LKV;
      b00[s] = kb0[kk]; b10[s] = kb1[kk]; b01[s] = kb0[kk + 16]; b11[s] = kb1[kk + 16];
    }
    v8f s0 = {}, s1 = {};
    s0 = wmma_bf16(aql.v, b00, s0);
    s0 = wmma_bf16(aqh.v, b10, s0);
    s1 = wmma_bf16(aql.v, b01, s1);
    s1 = wmma_bf16(aqh.v, b11, s1);

    // online softmax over 32 new columns; row i lives in VGPR i across 16 lanes of this half
    float corr[8], p0[8], p1[8];
    #pragma unroll
    for (int i = 0; i < 8; ++i) {
      float a0 = s0[i] * 0.125f;     // 1/sqrt(64)
      float a1 = s1[i] * 0.125f;
      float t = fmaxf(a0, a1);
      t = fmaxf(t, __shfl_xor(t, 1, 32));
      t = fmaxf(t, __shfl_xor(t, 2, 32));
      t = fmaxf(t, __shfl_xor(t, 4, 32));
      t = fmaxf(t, __shfl_xor(t, 8, 32));
      float mnew = fmaxf(mrun[i], t);
      corr[i] = __expf(mrun[i] - mnew);
      p0[i] = __expf(a0 - mnew);
      p1[i] = __expf(a1 - mnew);
      float r = p0[i] + p1[i];
      r += __shfl_xor(r, 1, 32);
      r += __shfl_xor(r, 2, 32);
      r += __shfl_xor(r, 4, 32);
      r += __shfl_xor(r, 8, 32);
      lrun[i] = lrun[i] * corr[i] + r;
      mrun[i] = mnew;
    }
    #pragma unroll
    for (int j = 0; j < 4; ++j)
      #pragma unroll
      for (int i = 0; i < 8; ++i) acc[j][i] *= corr[i];

    // P (C layout) -> LDS row-major [16][32] -> reload in A layout (same-wave LDS is in-order)
    #pragma unroll
    for (int i = 0; i < 8; ++i) {
      myp[(base + i) * 32 + mrow]      = (bf16)p0[i];
      myp[(base + i) * 32 + 16 + mrow] = (bf16)p1[i];
    }
    __asm volatile("" ::: "memory");
    A16 pa;
    pa.h[0] = *(const v8bf*)(myp + mrow * 32 + 8*half);
    pa.h[1] = *(const v8bf*)(myp + mrow * 32 + 16 + 8*half);
    __asm volatile("" ::: "memory");

    // out += P(16x32) @ V(32x64): B column d = mrow, elems = keys key0+k (table order)
    const bf16* vrow = vhead + (size_t)key0 * DMODEL + mrow;
    #pragma unroll
    for (int j = 0; j < 4; ++j) {
      v16bf vb;
      #pragma unroll
      for (int s = 0; s < 16; ++s) vb[s] = vrow[(size_t)((s & 7) + 8*half + 16*(s >> 3)) * DMODEL + 16*j];
      acc[j] = wmma_bf16(pa.v, vb, acc[j]);
    }
  }

  // normalize and write bf16 head output into attn_out [b][q][h*64+d]
  size_t orow0 = ((size_t)(b * LQ) + qt * 16) * DMODEL + h * HD;
  #pragma unroll
  for (int j = 0; j < 4; ++j) {
    int dcol = 16*j + mrow;
    #pragma unroll
    for (int i = 0; i < 8; ++i)
      outp[orow0 + (size_t)(base + i) * DMODEL + dcol] = (bf16)(acc[j][i] / lrun[i]);
  }
}

extern "C" void kernel_launch(void* const* d_in, const int* in_sizes, int n_in,
                              void* d_out, int out_size, void* d_ws, size_t ws_size,
                              hipStream_t stream) {
  const float* query = (const float*)d_in[0];
  const float* keyv  = (const float*)d_in[1];
  const float* qg = (const float*)d_in[2];
  const float* qb = (const float*)d_in[3];
  const float* kg = (const float*)d_in[4];
  const float* kb = (const float*)d_in[5];
  const float* vg = (const float*)d_in[6];
  const float* vb = (const float*)d_in[7];
  const float* Wq = (const float*)d_in[8];
  const float* bq = (const float*)d_in[9];
  const float* Wk = (const float*)d_in[10];
  const float* bk = (const float*)d_in[11];
  const float* Wv = (const float*)d_in[12];
  const float* bv = (const float*)d_in[13];
  const float* Wo = (const float*)d_in[14];
  const float* bo = (const float*)d_in[15];
  float* out = (float*)d_out;

  char* ws = (char*)d_ws;
  size_t off = 0;
  auto carve = [&](size_t elems) { bf16* p = (bf16*)(ws + off); off += ((elems * 2) + 255) & ~(size_t)255; return p; };
  bf16* qln   = carve((size_t)MQ * DMODEL);
  bf16* kln   = carve((size_t)MKV * DMODEL);
  bf16* vln   = carve((size_t)MKV * DMODEL);
  bf16* WqT   = carve((size_t)DMODEL * DMODEL);
  bf16* WkT   = carve((size_t)DMODEL * DMODEL);
  bf16* WvT   = carve((size_t)DMODEL * DMODEL);
  bf16* WoT   = carve((size_t)DMODEL * DMODEL);
  bf16* qproj = carve((size_t)MQ * DMODEL);
  bf16* kTb   = carve((size_t)MKV * DMODEL);   // [b][h][d][key]
  bf16* vproj = carve((size_t)MKV * DMODEL);
  bf16* aout  = carve((size_t)MQ * DMODEL);

  // LayerNorms (bf16 activations for WMMA)
  ln_kernel<<<MQ / 8, 256, 0, stream>>>(query, qg, qb, qln, MQ);
  ln_kernel<<<MKV / 8, 256, 0, stream>>>(keyv, kg, kb, kln, MKV);
  ln_kernel<<<MKV / 8, 256, 0, stream>>>(keyv, vg, vb, vln, MKV);

  // weight transposes (K-major bf16 for the B operand)
  wtrans_kernel<<<1024, 256, 0, stream>>>(Wq, WqT);
  wtrans_kernel<<<1024, 256, 0, stream>>>(Wk, WkT);
  wtrans_kernel<<<1024, 256, 0, stream>>>(Wv, WvT);
  wtrans_kernel<<<1024, 256, 0, stream>>>(Wo, WoT);

  // projections: Q natural, K directly in per-head kT layout, V natural
  gemm_kernel<<<(MQ / 16) * 16 / 8, 256, 0, stream>>>(qln, WqT, bq, MQ, 0, qproj, nullptr, nullptr);
  gemm_kernel<<<(MKV / 16) * 16 / 8, 256, 0, stream>>>(kln, WkT, bk, MKV, 1, kTb, nullptr, nullptr);
  gemm_kernel<<<(MKV / 16) * 16 / 8, 256, 0, stream>>>(vln, WvT, bv, MKV, 0, vproj, nullptr, nullptr);

  // flash attention: one wave per (b, h, 16-query tile) = 4096 waves
  attn_kernel<<<(NB * NHEAD * (LQ / 16)) / 8, 256, 0, stream>>>(qproj, kTb, vproj, aout);

  // output projection + bias + residual, f32 out
  gemm_kernel<<<(MQ / 16) * 16 / 8, 256, 0, stream>>>(aout, WoT, bo, MQ, 2, nullptr, out, query);
}
